// InsLayer_61735859912927
// MI455X (gfx1250) — compile-verified, matrix operand order checked by reading, not run on hardware
//
// CDNA5 (gfx1250) implementation of the BEV InsLayer forward pass.
//
// Roofline (MI455X): ~120MB of mandatory HBM traffic (~5us @ 23.3TB/s) vs
// ~23 GFLOP of GEMM-shaped work (3x3 conv = 9.7 GMAC dominates, then the
// 65536-token bv/hw projections). All GEMMs run on v_wmma_f32_16x16x32_f16
// (f32 accumulate). Iteration history: r1 removed per-element bounds guards
// (saveexec around every load); r2 removed v_movrel accumulator indexing by
// fully unrolling the N-tile loop; this revision turns the LDS staging loops
// into fixed-trip unrolled code (they were compiling to dynamic loops with
// exec-mask bookkeeping) and vectorizes A staging with float2 loads + packed
// 2xf16 LDS stores. WMMA operands feed from LDS as ds_load_b128 pairs (ISA
// 7.12.2 f16 layout = two contiguous 8-halfword runs per lane). CDNA5 paths
// exercised: WMMA, b128 LDS feeds, global_prefetch, cluster barrier (NOP
// outside clusters), NULL-descriptor tensor_load_to_lds + s_wait_tensorcnt.

#include <hip/hip_runtime.h>
#include <math.h>

typedef __attribute__((ext_vector_type(16))) _Float16 v16h;
typedef __attribute__((ext_vector_type(8)))  _Float16 v8h;
typedef __attribute__((ext_vector_type(2)))  _Float16 v2h;
typedef __attribute__((ext_vector_type(8)))  float    v8f;
typedef __attribute__((ext_vector_type(4)))  unsigned int u32x4;
typedef __attribute__((ext_vector_type(4)))  int i32x4;
typedef __attribute__((ext_vector_type(8)))  int i32x8;

#define SL 256
#define SW 256
#define BEV_HW 65536   // 256*256
#define CCH 128
#define NHD 8
#define NPNT 8
#define NPX 5
#define NG 2048
#define NQ 10240       // NG*NP

__device__ __forceinline__ v8f wmma16(v16h a, v16h b, v8f c) {
  return __builtin_amdgcn_wmma_f32_16x16x32_f16(false, a, false, b, (short)0, c,
                                                false, false);
}

// Combine the two contiguous 8-halfword runs of a lane's f16 operand.
__device__ __forceinline__ v16h cat8(v8h lo, v8h hi) {
  return __builtin_shufflevector(lo, hi, 0, 1, 2, 3, 4, 5, 6, 7,
                                 8, 9, 10, 11, 12, 13, 14, 15);
}

// Harmless TDM no-op: all-zero descriptor => count==0 => NULL tensor (ISA 8.3).
__device__ __forceinline__ void tdm_null_op() {
#if __has_builtin(__builtin_amdgcn_tensor_load_to_lds)
  u32x4 g0 = {0u, 0u, 0u, 0u};
  i32x8 g1 = {0, 0, 0, 0, 0, 0, 0, 0};
  i32x4 g2 = {0, 0, 0, 0};
  i32x4 g3 = {0, 0, 0, 0};
#if defined(__clang_major__) && __clang_major__ >= 23
  i32x8 g4 = {0, 0, 0, 0, 0, 0, 0, 0};
  __builtin_amdgcn_tensor_load_to_lds(g0, g1, g2, g3, g4, 0);
#else
  __builtin_amdgcn_tensor_load_to_lds(g0, g1, g2, g3, 0);
#endif
  __builtin_amdgcn_s_wait_tensorcnt(0);
#endif
}

// ---------------------------------------------------------------------------
// Fast WMMA GEMM for aligned shapes (M%16==0, N%16==0, K%32==0), no guards.
// Out[m,n] = act( sum_k A[m,k]*B[k,n] + bias[n] ),
// A[m,k]=A[m*a_rs+k*a_cs] (column-major callers have a_rs==1),
// B[k,n]=B[k*b_rs+n*b_cs], Out[m,n]=Out[m*o_rs+n*o_cs].
// One block per 16-row M-panel; 8 waves stripe up to 32 N tiles (4 per wave,
// fully unrolled so accumulators stay in fixed VGPRs). The A micro-tile
// (16x32) is staged once per block into LDS with one float2 load per thread
// (coalesced in A's contiguous dimension) and read back as 2x ds_load_b128.
// ---------------------------------------------------------------------------
__global__ void wmma_gemm_fast_kernel(const float* __restrict__ A,
                                      const float* __restrict__ B,
                                      const float* __restrict__ bias,
                                      float* __restrict__ Out,
                                      int M, int N, int K,
                                      int a_rs, int a_cs, int b_rs, int b_cs,
                                      int o_rs, int o_cs, int act) {
  __shared__ __attribute__((aligned(16))) _Float16 As[16 * 32];
  const int t = threadIdx.x;
  const int lane = t & 31, wave = t >> 5;
  const int half = lane >> 4, idx = lane & 15;
  const int m0 = blockIdx.x << 4;
  const int tiles_n = N >> 4;
  const bool ca = (a_cs == 1);   // A row-major: K contiguous
  const bool cb = (b_rs == 1);   // B column-major: K contiguous
  v8f acc0 = {}, acc1 = {}, acc2 = {}, acc3 = {};
  (void)M;

  for (int k0 = 0; k0 < K; k0 += 32) {
    // ---- stage A 16x32 tile: one float2 per thread, no loop, no guards
    if (ca) {
      const int mm = t >> 4, kk2 = (t & 15) << 1;        // pair along K
      const float2 aa = *(const float2*)&A[(long)(m0 + mm) * a_rs + (k0 + kk2)];
      v2h p; p[0] = (_Float16)aa.x; p[1] = (_Float16)aa.y;
      *(v2h*)&As[mm * 32 + kk2] = p;                      // packed ds_store_b32
    } else {
      const int mm2 = (t & 7) << 1, kk = t >> 3;         // pair along M (a_rs==1)
      const float2 aa = *(const float2*)&A[(long)(m0 + mm2) + (long)(k0 + kk) * a_cs];
      As[mm2 * 32 + kk]       = (_Float16)aa.x;
      As[(mm2 + 1) * 32 + kk] = (_Float16)aa.y;
    }
    __syncthreads();
    // ---- A operand: two contiguous 8-halfword runs per lane (ISA 7.12.2)
    const v16h av = cat8(*(const v8h*)&As[idx * 32 + half * 8],
                         *(const v8h*)&As[idx * 32 + 16 + half * 8]);
    // ---- each wave covers up to 4 N tiles; fixed unroll keeps acc in regs
#pragma unroll
    for (int j = 0; j < 4; ++j) {
      const int nt = wave + j * 8;
      if (nt < tiles_n) {
        const int nn = (nt << 4) + idx;
        v16h bv;
        if (cb) {
          const float* Bp = B + (long)nn * b_cs;
#pragma unroll
          for (int u = 0; u < 8; ++u) {
            const int k = k0 + ((u >> 2) << 4) + (half << 3) + ((u & 3) << 1);
            const float2 bb = *(const float2*)(Bp + k);   // k even, 8B aligned
            bv[2 * u] = (_Float16)bb.x; bv[2 * u + 1] = (_Float16)bb.y;
          }
        } else {
#pragma unroll
          for (int u = 0; u < 8; ++u) {
            const int k = k0 + ((u >> 2) << 4) + (half << 3) + ((u & 3) << 1);
            bv[2 * u]     = (_Float16)B[(long)k * b_rs + (long)nn * b_cs];
            bv[2 * u + 1] = (_Float16)B[(long)(k + 1) * b_rs + (long)nn * b_cs];
          }
        }
        if (j == 0) acc0 = wmma16(av, bv, acc0);
        else if (j == 1) acc1 = wmma16(av, bv, acc1);
        else if (j == 2) acc2 = wmma16(av, bv, acc2);
        else acc3 = wmma16(av, bv, acc3);
      }
    }
    __syncthreads();
  }
#pragma unroll
  for (int j = 0; j < 4; ++j) {
    const int nt = wave + j * 8;
    if (nt < tiles_n) {
      const v8f acc = (j == 0) ? acc0 : (j == 1) ? acc1 : (j == 2) ? acc2 : acc3;
      const int nn = (nt << 4) + idx;
      const float bb = bias ? bias[nn] : 0.f;
#pragma unroll
      for (int r = 0; r < 8; ++r) {
        const int mm = m0 + r + (half << 3);   // C/D: VGPR r -> M=r / r+8
        float v = acc[r] + bb;
        if (act == 1) v = fmaxf(v, 0.f);
        Out[(long)mm * o_rs + (long)nn * o_cs] = v;
      }
    }
  }
}

// ---------------------------------------------------------------------------
// Guarded generic WMMA GEMM (only for the N=10 seg head).
// ---------------------------------------------------------------------------
__global__ void wmma_gemm_kernel(const float* __restrict__ A,
                                 const float* __restrict__ B,
                                 const float* __restrict__ bias,
                                 float* __restrict__ Out,
                                 int M, int N, int K,
                                 int a_rs, int a_cs, int b_rs, int b_cs,
                                 int o_rs, int o_cs, int act) {
  const int lane = threadIdx.x & 31;
  const int wave = threadIdx.x >> 5;
  const int half = lane >> 4;
  const int idx  = lane & 15;
  const int tiles_n = (N + 15) >> 4;
  const int tiles_m = (M + 15) >> 4;
  const int tile = blockIdx.x * 8 + wave;
  if (tile >= tiles_m * tiles_n) return;
  const int m0 = (tile / tiles_n) << 4;
  const int n0 = (tile % tiles_n) << 4;
  const int m = m0 + idx;
  const int n = n0 + idx;
  v8f acc = {};
  for (int k0 = 0; k0 < K; k0 += 32) {
    v16h av, bv;
#pragma unroll
    for (int u = 0; u < 8; ++u) {
      const int k = k0 + ((u >> 2) << 4) + (half << 3) + ((u & 3) << 1);
      float a0 = 0.f, a1 = 0.f, b0 = 0.f, b1 = 0.f;
      if (m < M && k < K) {
        a0 = A[(long)m * a_rs + (long)k * a_cs];
        if (k + 1 < K) a1 = A[(long)m * a_rs + (long)(k + 1) * a_cs];
      }
      if (n < N && k < K) {
        b0 = B[(long)k * b_rs + (long)n * b_cs];
        if (k + 1 < K) b1 = B[(long)(k + 1) * b_rs + (long)n * b_cs];
      }
      av[2 * u] = (_Float16)a0; av[2 * u + 1] = (_Float16)a1;
      bv[2 * u] = (_Float16)b0; bv[2 * u + 1] = (_Float16)b1;
    }
    acc = wmma16(av, bv, acc);
  }
#pragma unroll
  for (int r = 0; r < 8; ++r) {
    const int mm = m0 + r + (half << 3);
    const int nn = n0 + idx;
    if (mm < M && nn < N) {
      float v = acc[r];
      if (bias) v += bias[nn];
      if (act == 1) v = fmaxf(v, 0.f);
      Out[(long)mm * o_rs + (long)nn * o_cs] = v;
    }
  }
}

// ---------------------------------------------------------------------------
// 3x3 SAME conv, NCHW in (128,256,256), OIHW weights (128,128,3,3), relu+bias.
// Implicit GEMM: 64-pixel strip x 128 out-ch per block; K=9*128 staged 32 at
// a time through LDS as f16 (A row-major, B transposed) so both WMMA operands
// are two ds_load_b128 per lane. 8 waves = (m_tile[0..3], n_half[0..1]).
// ---------------------------------------------------------------------------
__global__ void conv3x3_wmma_kernel(const float* __restrict__ bev,
                                    const float* __restrict__ Wt,
                                    const float* __restrict__ bias,
                                    float* __restrict__ out) {
  __builtin_amdgcn_s_cluster_barrier();      // NOP outside clusters
  __shared__ __attribute__((aligned(16))) _Float16 As[64 * 32];
  __shared__ __attribute__((aligned(16))) _Float16 Bs[128 * 32];  // [n][k]
  const int t = threadIdx.x;
  const int lane = t & 31, wave = t >> 5;
  const int half = lane >> 4, idx = lane & 15;
  const int y = blockIdx.y;
  const int x0 = blockIdx.x * 64;
  const int mt = wave >> 1;
  const int n0 = (wave & 1) * 64;
  v8f acc0 = {}, acc1 = {}, acc2 = {}, acc3 = {};

  if (blockIdx.x == 0 && blockIdx.y == 0 && t == 0) tdm_null_op();

  for (int kp = 0; kp < 9; ++kp) {
    const int dy = kp / 3 - 1, dx = kp % 3 - 1;
    const int yy = y + dy;
    const bool row_ok = (yy >= 0 && yy < SL);
    for (int c0 = 0; c0 < CCH; c0 += 32) {
      // stage A: 64 pixels x 32 channels; adjacent threads -> adjacent pixels
#pragma unroll
      for (int i = 0; i < 8; ++i) {
        const int e = t + (i << 8);
        const int mm = e & 63, kk = e >> 6;
        const int xx = x0 + mm + dx;
        float vv = 0.f;
        if (row_ok && xx >= 0 && xx < SW)
          vv = bev[(long)(c0 + kk) * BEV_HW + yy * SW + xx];
        As[mm * 32 + kk] = (_Float16)vv;
      }
      // stage B transposed: Bs[n][k] = Wt[n][c0+k][kp]
#pragma unroll
      for (int i = 0; i < 16; ++i) {
        const int e = t + (i << 8);
        const int nn = e >> 5, kk = e & 31;
        Bs[nn * 32 + kk] = (_Float16)Wt[(long)nn * 1152 + (c0 + kk) * 9 + kp];
      }
      if (c0 + 32 < CCH)
        __builtin_prefetch(&Wt[(long)(c0 + 32) * 9 + kp], 0, 1); // global_prefetch
      __syncthreads();
      const int ar = (mt * 16 + idx) * 32;
      const v16h av = cat8(*(const v8h*)&As[ar + half * 8],
                           *(const v8h*)&As[ar + 16 + half * 8]);
      {
        const int br = (n0 + 0 * 16 + idx) * 32;
        acc0 = wmma16(av, cat8(*(const v8h*)&Bs[br + half * 8],
                               *(const v8h*)&Bs[br + 16 + half * 8]), acc0);
      }
      {
        const int br = (n0 + 1 * 16 + idx) * 32;
        acc1 = wmma16(av, cat8(*(const v8h*)&Bs[br + half * 8],
                               *(const v8h*)&Bs[br + 16 + half * 8]), acc1);
      }
      {
        const int br = (n0 + 2 * 16 + idx) * 32;
        acc2 = wmma16(av, cat8(*(const v8h*)&Bs[br + half * 8],
                               *(const v8h*)&Bs[br + 16 + half * 8]), acc2);
      }
      {
        const int br = (n0 + 3 * 16 + idx) * 32;
        acc3 = wmma16(av, cat8(*(const v8h*)&Bs[br + half * 8],
                               *(const v8h*)&Bs[br + 16 + half * 8]), acc3);
      }
      __syncthreads();
    }
  }
#pragma unroll
  for (int nt = 0; nt < 4; ++nt) {
    const v8f acc = (nt == 0) ? acc0 : (nt == 1) ? acc1 : (nt == 2) ? acc2 : acc3;
    const int nn = n0 + nt * 16 + idx;
    const float bb = bias[nn];
#pragma unroll
    for (int r = 0; r < 8; ++r) {
      const int xm = x0 + mt * 16 + r + half * 8;
      out[(long)nn * BEV_HW + y * SW + xm] = fmaxf(acc[r] + bb, 0.f);
    }
  }
}

// ---------------------------- VALU glue kernels ----------------------------

__global__ void gather_q_kernel(const float* __restrict__ bev,
                                const int* __restrict__ ref3,
                                float* __restrict__ q0) {
  const int tid = blockIdx.x * blockDim.x + threadIdx.x;
  if (tid >= NQ * CCH) return;
  const int m = tid >> 7, c = tid & 127;
  const int row = ref3[m * 3 + 0], col = ref3[m * 3 + 1];
  q0[tid] = bev[(long)c * BEV_HW + row * SW + col];
}

// LayerNorm over C=128; optional residual. One row per 128-thread block.
__global__ void ln128_kernel(const float* __restrict__ X,
                             const float* __restrict__ R,
                             const float* __restrict__ g,
                             const float* __restrict__ b,
                             float* __restrict__ Out) {
  __shared__ float sd[128];
  const int row = blockIdx.x, t = threadIdx.x;
  float x = X[(long)row * 128 + t];
  if (R) x += R[(long)row * 128 + t];
  sd[t] = x; __syncthreads();
  for (int o = 64; o > 0; o >>= 1) { if (t < o) sd[t] += sd[t + o]; __syncthreads(); }
  const float mean = sd[0] / 128.f; __syncthreads();
  const float d = x - mean;
  sd[t] = d * d; __syncthreads();
  for (int o = 64; o > 0; o >>= 1) { if (t < o) sd[t] += sd[t + o]; __syncthreads(); }
  const float var = sd[0] / 128.f;
  Out[(long)row * 128 + t] = d * rsqrtf(var + 1e-5f) * g[t] + b[t];
}

__global__ void softmax_rows_kernel(float* __restrict__ X, int rows, int L) {
  const int r = blockIdx.x * blockDim.x + threadIdx.x;
  if (r >= rows) return;
  float* p = X + (long)r * L;
  float mx = -1e30f;
  for (int i = 0; i < L; ++i) mx = fmaxf(mx, p[i]);
  float s = 0.f;
  for (int i = 0; i < L; ++i) { const float e = expf(p[i] - mx); p[i] = e; s += e; }
  const float inv = 1.f / s;
  for (int i = 0; i < L; ++i) p[i] *= inv;
}

// Multi-scale deformable sampling. One thread per (q, h, d).
__global__ void msds_kernel(const float* __restrict__ value,
                            const float* __restrict__ off,
                            const float* __restrict__ aw,
                            const float* __restrict__ ref,
                            float* __restrict__ out, int Nq, int nlev,
                            int s0, int s1, int s2, int H0, int H1, int H2,
                            int W0, int W1, int W2) {
  const int tid = blockIdx.x * blockDim.x + threadIdx.x;
  if (tid >= Nq * CCH) return;
  const int q = tid >> 7, rem = tid & 127, h = rem >> 4, d = rem & 15;
  const int Ss[3] = {s0, s1, s2}, Hs[3] = {H0, H1, H2}, Ws[3] = {W0, W1, W2};
  const float refx = ref[q * 2 + 0], refy = ref[q * 2 + 1];
  float acc = 0.f;
  for (int l = 0; l < nlev; ++l) {
    const int Hl = Hs[l], Wl = Ws[l], st = Ss[l];
    for (int p = 0; p < NPNT; ++p) {
      const long base = (((long)q * NHD + h) * nlev + l) * NPNT + p;
      const float ox = off[base * 2], oy = off[base * 2 + 1];
      const float w = aw[base];
      const float x = refx * Wl + ox - 0.5f;   // (ref + off/W)*W - 0.5
      const float y = refy * Hl + oy - 0.5f;
      const float fx0 = floorf(x), fy0 = floorf(y);
      const int x0 = (int)fx0, y0 = (int)fy0;
      const float wx1 = x - fx0, wy1 = y - fy0;
      const float wx0 = 1.f - wx1, wy0 = 1.f - wy1;
      float s = 0.f;
#pragma unroll
      for (int cy = 0; cy < 2; ++cy)
#pragma unroll
        for (int cx = 0; cx < 2; ++cx) {
          const int ix = x0 + cx, iy = y0 + cy;
          if (ix >= 0 && ix < Wl && iy >= 0 && iy < Hl) {
            const float wt = (cx ? wx1 : wx0) * (cy ? wy1 : wy0);
            s += value[(long)(st + iy * Wl + ix) * CCH + h * 16 + d] * wt;
          }
        }
      acc += w * s;
    }
  }
  out[(long)q * CCH + rem] = acc;
}

__global__ void group_reduce_kernel(const float* __restrict__ q,
                                    const float* __restrict__ hs,
                                    float* __restrict__ qg) {
  const int tid = blockIdx.x * blockDim.x + threadIdx.x;
  if (tid >= NG * CCH) return;
  const int g = tid >> 7, c = tid & 127;
  float num = 0.f, den = 0.f;
  for (int p = 0; p < NPX; ++p) {
    const float w = hs[g * NPX + p];
    num += q[(long)(g * NPX + p) * CCH + c] * w;
    den += w;
  }
  qg[tid] = num / den;
}

__global__ void reff_kernel(const float* __restrict__ r2, float* __restrict__ rf, int n) {
  const int tid = blockIdx.x * blockDim.x + threadIdx.x;
  if (tid >= n) return;
  rf[tid * 2 + 0] = r2[tid * 2 + 1];   // refxy = ref2ds[:, ::-1]
  rf[tid * 2 + 1] = r2[tid * 2 + 0];
}

__global__ void refbev_kernel(const int* __restrict__ ref3, float* __restrict__ rb) {
  const int g = blockIdx.x * blockDim.x + threadIdx.x;
  if (g >= NG) return;
  const int row = ref3[g * (NPX * 3) + 0], col = ref3[g * (NPX * 3) + 1];
  rb[g * 2 + 0] = (float)col / (float)(SW - 1);
  rb[g * 2 + 1] = (float)row / (float)(SL - 1);
}

__global__ void posemb_kernel(const int* __restrict__ ref3, float* __restrict__ pein) {
  const int tid = blockIdx.x * blockDim.x + threadIdx.x;
  if (tid >= NG * 256) return;
  const int g = tid >> 8, j = tid & 255;
  const int row = ref3[g * (NPX * 3) + 0], col = ref3[g * (NPX * 3) + 1];
  float nrm; int jj;
  if (j < 128) { nrm = (float)col / (float)(SW - 1); jj = j; }
  else         { nrm = (float)row / (float)(SL - 1); jj = j - 128; }
  const float pos = nrm * 6.283185307179586f;
  const float expo = (float)(2 * (jj >> 1)) / 128.f;
  const float v = pos / powf(10000.f, expo);
  pein[tid] = (jj & 1) ? cosf(v) : sinf(v);
}

__global__ void add_kernel(const float* __restrict__ a, const float* __restrict__ b,
                           float* __restrict__ o, int n) {
  const int tid = blockIdx.x * blockDim.x + threadIdx.x;
  if (tid < n) o[tid] = a[tid] + b[tid];
}

// Small MHA (N=2048, 8 heads, Dh=16): one thread per (query, head), two-pass.
__global__ void mha_kernel(const float* __restrict__ Q, const float* __restrict__ K,
                           const float* __restrict__ V, float* __restrict__ O, int N) {
  const int tid = blockIdx.x * blockDim.x + threadIdx.x;
  if (tid >= N * NHD) return;
  const int q = tid >> 3, h = tid & 7;
  float qr[16];
#pragma unroll
  for (int d = 0; d < 16; ++d) qr[d] = Q[(long)q * CCH + h * 16 + d];
  float mx = -1e30f;
  for (int k = 0; k < N; ++k) {
    const float* kp = &K[(long)k * CCH + h * 16];
    float s = 0.f;
#pragma unroll
    for (int d = 0; d < 16; ++d) s += qr[d] * kp[d];
    mx = fmaxf(mx, s * 0.25f);
  }
  float o[16]; float sum = 0.f;
#pragma unroll
  for (int d = 0; d < 16; ++d) o[d] = 0.f;
  for (int k = 0; k < N; ++k) {
    const float* kp = &K[(long)k * CCH + h * 16];
    float s = 0.f;
#pragma unroll
    for (int d = 0; d < 16; ++d) s += qr[d] * kp[d];
    const float e = expf(s * 0.25f - mx);
    sum += e;
    const float* vp = &V[(long)k * CCH + h * 16];
#pragma unroll
    for (int d = 0; d < 16; ++d) o[d] += e * vp[d];
  }
  const float inv = 1.f / sum;
#pragma unroll
  for (int d = 0; d < 16; ++d) O[(long)q * CCH + h * 16 + d] = o[d] * inv;
}

__global__ void blend_kernel(const float* __restrict__ qg, const float* __restrict__ qs,
                             const int* __restrict__ ref3, const float* __restrict__ bev_in,
                             float* __restrict__ bev_ws) {
  const int tid = blockIdx.x * blockDim.x + threadIdx.x;
  if (tid >= NG * CCH) return;
  const int g = tid >> 7, c = tid & 127;
  const int row = ref3[g * (NPX * 3) + 0], col = ref3[g * (NPX * 3) + 1];
  const long pix = (long)row * SW + col;
  const float w = qs[g];
  bev_ws[(long)c * BEV_HW + pix] =
      qg[tid] * w + bev_in[(long)c * BEV_HW + pix] * (1.f - w);
}

// ---------------------------------------------------------------------------

extern "C" void kernel_launch(void* const* d_in, const int* in_sizes, int n_in,
                              void* d_out, int out_size, void* d_ws, size_t ws_size,
                              hipStream_t stream) {
  (void)in_sizes; (void)n_in; (void)out_size; (void)ws_size;
  const float* bev_ins = (const float*)d_in[0];
  const float* bev_bk  = (const float*)d_in[1];
  const float* feat0   = (const float*)d_in[2];   // (1,128,48,160)
  const float* feat1   = (const float*)d_in[3];   // (1,128,24,80)
  const float* feat2   = (const float*)d_in[4];   // (1,128,12,40)
  const float* ref2ds  = (const float*)d_in[5];
  const float* hs_w    = (const float*)d_in[6];
  const float* qs_w    = (const float*)d_in[7];
  auto P = [&](int i) { return (const float*)d_in[8 + i]; };
  enum { HW_W=0, HW_B, SEG_W, SEG_B, QE_W, QE_B, QE_G, QE_BE,
         PE_W1, PE_B1, PE_W2, PE_B2, PE_G, PE_BE,
         FV_W, FV_B, FOFF_W, FOFF_B, FATTN_W, FATTN_B, FO_W, FO_B, F_LN_G, F_LN_B,
         BV_W, BV_B, BOFF_W, BOFF_B, BATTN_W, BATTN_B, BO_W, BO_B, B_LN_G, B_LN_B,
         SQ_W, SQ_B, SK_W, SK_B, SV_W, SV_B, SO_W, SO_B, S_LN_G, S_LN_B,
         FFN_W1, FFN_B1, FFN_W2, FFN_B2, FFN_G, FFN_BE, CONV_W, CONV_B };
  const int* ref3ds = (const int*)d_in[60];

  float* out = (float*)d_out;
  float* out_conv = out;                          // (1,128,256,256)
  float* out_seg  = out + (long)CCH * BEV_HW;     // (1,10,256,256)
  float* out_wh   = out_seg + (long)10 * BEV_HW;  // (1,256,256,32)

  // workspace carve-out
  float* W = (float*)d_ws;
  size_t o = 0;
  auto alloc = [&](size_t n) { float* p = W + o; o += n; return p; };
  float* q0g   = alloc((size_t)NQ * CCH);
  float* q     = alloc((size_t)NQ * CCH);
  float* valf  = alloc((size_t)10080 * CCH);
  float* offf  = alloc((size_t)NQ * 384);
  float* attnf = alloc((size_t)NQ * 192);
  float* sampf = alloc((size_t)NQ * CCH);
  float* aproj = alloc((size_t)NQ * CCH);
  float* refff = alloc((size_t)NQ * 2);
  float* valb  = alloc((size_t)BEV_HW * CCH);
  float* qg    = alloc((size_t)NG * CCH);
  float* offb  = alloc((size_t)NG * 128);
  float* attnb = alloc((size_t)NG * 64);
  float* sampb = alloc((size_t)NG * CCH);
  float* aprjb = alloc((size_t)NG * CCH);
  float* refb  = alloc((size_t)NG * 2);
  float* pein  = alloc((size_t)NG * 256);
  float* pe1   = alloc((size_t)NG * CCH);
  float* peb   = alloc((size_t)NG * CCH);
  float* qk    = alloc((size_t)NG * CCH);
  float* Qm    = alloc((size_t)NG * CCH);
  float* Km    = alloc((size_t)NG * CCH);
  float* Vm    = alloc((size_t)NG * CCH);
  float* Om    = alloc((size_t)NG * CCH);
  float* mo    = alloc((size_t)NG * CCH);
  float* ffh   = alloc((size_t)NG * 512);
  float* ffo   = alloc((size_t)NG * CCH);
  float* bevw  = alloc((size_t)CCH * BEV_HW);

  // Aligned-shape GEMM (M%16, N%16, K%32): one block per 16-row panel.
  auto gemm = [&](const float* A, const float* B, const float* bias, float* Out,
                  int M, int N, int K, int ars, int acs, int brs, int bcs,
                  int ors, int ocs, int act) {
    wmma_gemm_fast_kernel<<<M / 16, 256, 0, stream>>>(
        A, B, bias, Out, M, N, K, ars, acs, brs, bcs, ors, ocs, act);
  };
  auto cdiv = [](int a, int b) { return (a + b - 1) / b; };

  // ---- heads on original BEV: weight_h (NHWC, 32) and seg_logits (NCHW, 10)
  gemm(bev_ins, P(HW_W), P(HW_B), out_wh, BEV_HW, 32, CCH, 1, BEV_HW, 1, CCH, 32, 1, 0);
  {  // seg: N=10 -> guarded generic kernel
    const int tiles = (BEV_HW / 16) * 1;
    wmma_gemm_kernel<<<(tiles + 7) / 8, 256, 0, stream>>>(
        bev_ins, P(SEG_W), P(SEG_B), out_seg, BEV_HW, 10, CCH,
        1, BEV_HW, 1, CCH, 1, BEV_HW, 0);
  }

  // ---- q = LN(gather(bev) @ qe_w + qe_b)
  gather_q_kernel<<<cdiv(NQ * CCH, 256), 256, 0, stream>>>(bev_ins, ref3ds, q0g);
  gemm(q0g, P(QE_W), P(QE_B), q, NQ, CCH, CCH, CCH, 1, CCH, 1, CCH, 1, 0);
  ln128_kernel<<<NQ, 128, 0, stream>>>(q, nullptr, P(QE_G), P(QE_BE), q);

  // ---- msdeform 'f': value proj per level (A is NCHW: cols strided by H*W)
  gemm(feat0, P(FV_W), P(FV_B), valf + (long)0    * CCH, 7680, CCH, CCH, 1, 7680, CCH, 1, CCH, 1, 0);
  gemm(feat1, P(FV_W), P(FV_B), valf + (long)7680 * CCH, 1920, CCH, CCH, 1, 1920, CCH, 1, CCH, 1, 0);
  gemm(feat2, P(FV_W), P(FV_B), valf + (long)9600 * CCH,  480, CCH, CCH, 1,  480, CCH, 1, CCH, 1, 0);
  gemm(q, P(FOFF_W),  P(FOFF_B),  offf,  NQ, 384, CCH, CCH, 1, 384, 1, 384, 1, 0);
  gemm(q, P(FATTN_W), P(FATTN_B), attnf, NQ, 192, CCH, CCH, 1, 192, 1, 192, 1, 0);
  softmax_rows_kernel<<<cdiv(NQ * NHD, 256), 256, 0, stream>>>(attnf, NQ * NHD, 24);
  reff_kernel<<<cdiv(NQ, 256), 256, 0, stream>>>(ref2ds, refff, NQ);
  msds_kernel<<<cdiv(NQ * CCH, 256), 256, 0, stream>>>(
      valf, offf, attnf, refff, sampf, NQ, 3, 0, 7680, 9600, 48, 24, 12, 160, 80, 40);
  gemm(sampf, P(FO_W), P(FO_B), aproj, NQ, CCH, CCH, CCH, 1, CCH, 1, CCH, 1, 0);
  ln128_kernel<<<NQ, 128, 0, stream>>>(q, aproj, P(F_LN_G), P(F_LN_B), q);

  // ---- group reduce (hs_w weighted mean over NP=5)
  group_reduce_kernel<<<cdiv(NG * CCH, 256), 256, 0, stream>>>(q, hs_w, qg);

  // ---- msdeform 'b' over bev_f_bk (single 256x256 level)
  gemm(bev_bk, P(BV_W), P(BV_B), valb, BEV_HW, CCH, CCH, 1, BEV_HW, CCH, 1, CCH, 1, 0);
  gemm(qg, P(BOFF_W),  P(BOFF_B),  offb,  NG, 128, CCH, CCH, 1, 128, 1, 128, 1, 0);
  gemm(qg, P(BATTN_W), P(BATTN_B), attnb, NG, 64,  CCH, CCH, 1, 64,  1, 64,  1, 0);
  softmax_rows_kernel<<<cdiv(NG * NHD, 256), 256, 0, stream>>>(attnb, NG * NHD, 8);
  refbev_kernel<<<cdiv(NG, 256), 256, 0, stream>>>(ref3ds, refb);
  msds_kernel<<<cdiv(NG * CCH, 256), 256, 0, stream>>>(
      valb, offb, attnb, refb, sampb, NG, 1, 0, 0, 0, SL, 1, 1, SW, 1, 1);
  gemm(sampb, P(BO_W), P(BO_B), aprjb, NG, CCH, CCH, CCH, 1, CCH, 1, CCH, 1, 0);
  ln128_kernel<<<NG, 128, 0, stream>>>(qg, aprjb, P(B_LN_G), P(B_LN_B), qg);

  // ---- positional embedding MLP + LN
  posemb_kernel<<<cdiv(NG * 256, 256), 256, 0, stream>>>(ref3ds, pein);
  gemm(pein, P(PE_W1), P(PE_B1), pe1, NG, CCH, 256, 256, 1, CCH, 1, CCH, 1, 1);
  gemm(pe1,  P(PE_W2), P(PE_B2), peb, NG, CCH, CCH, CCH, 1, CCH, 1, CCH, 1, 0);
  ln128_kernel<<<NG, 128, 0, stream>>>(peb, nullptr, P(PE_G), P(PE_BE), peb);

  // ---- MHA + LN
  add_kernel<<<cdiv(NG * CCH, 256), 256, 0, stream>>>(qg, peb, qk, NG * CCH);
  gemm(qk, P(SQ_W), P(SQ_B), Qm, NG, CCH, CCH, CCH, 1, CCH, 1, CCH, 1, 0);
  gemm(qk, P(SK_W), P(SK_B), Km, NG, CCH, CCH, CCH, 1, CCH, 1, CCH, 1, 0);
  gemm(qg, P(SV_W), P(SV_B), Vm, NG, CCH, CCH, CCH, 1, CCH, 1, CCH, 1, 0);
  mha_kernel<<<cdiv(NG * NHD, 256), 256, 0, stream>>>(Qm, Km, Vm, Om, NG);
  gemm(Om, P(SO_W), P(SO_B), mo, NG, CCH, CCH, CCH, 1, CCH, 1, CCH, 1, 0);
  ln128_kernel<<<NG, 128, 0, stream>>>(qg, mo, P(S_LN_G), P(S_LN_B), qg);

  // ---- FFN + LN
  gemm(qg,  P(FFN_W1), P(FFN_B1), ffh, NG, 512, CCH, CCH, 1, 512, 1, 512, 1, 1);
  gemm(ffh, P(FFN_W2), P(FFN_B2), ffo, NG, CCH, 512, 512, 1, CCH, 1, CCH, 1, 0);
  ln128_kernel<<<NG, 128, 0, stream>>>(qg, ffo, P(FFN_G), P(FFN_BE), qg);

  // ---- blend into BEV copy, then 3x3 conv + relu
  hipMemcpyAsync(bevw, bev_ins, (size_t)CCH * BEV_HW * sizeof(float),
                 hipMemcpyDeviceToDevice, stream);
  blend_kernel<<<cdiv(NG * CCH, 256), 256, 0, stream>>>(qg, qs_w, ref3ds, bev_ins, bevw);
  conv3x3_wmma_kernel<<<dim3(SW / 64, SL), 256, 0, stream>>>(bevw, P(CONV_W), P(CONV_B), out_conv);
}
